// PCC_Layer_54537494724975
// MI455X (gfx1250) — matrix operand order repaired and checked by reading, not compiled, not run on hardware
//
#include <hip/hip_runtime.h>

#define N_ROWS   262144
#define N_CLASS  512
#define EMB_DIM  512
#define M_TILE   64                 // rows per block
#define ALD_STRIDE 520              // bf16 elems: 1040 B/row -> 4-bank shift, conflict-free

typedef __attribute__((ext_vector_type(16))) __bf16 v16bf;
typedef __attribute__((ext_vector_type(8)))  float  v8f;

static __device__ __forceinline__ unsigned short f32_to_bf16(float f) {
    union { float f; unsigned int u; } v; v.f = f;
    unsigned int u = v.u;
    unsigned int r = u + 0x7FFFu + ((u >> 16) & 1u);   // round-to-nearest-even
    return (unsigned short)(r >> 16);
}

// Pack two floats -> packed bf16x2 dword (hardware cvt if available).
static __device__ __forceinline__ unsigned int pack_bf16x2(float lo, float hi) {
#if __has_builtin(__builtin_amdgcn_cvt_pk_bf16_f32)
    typedef __attribute__((ext_vector_type(2))) __bf16 v2bf;
    union { v2bf v; unsigned int u; } r;
    r.v = __builtin_amdgcn_cvt_pk_bf16_f32(lo, hi);
    return r.u;
#else
    return (unsigned int)f32_to_bf16(lo) | ((unsigned int)f32_to_bf16(hi) << 16);
#endif
}

static __device__ __forceinline__ float fast_rcp(float x) {
#if __has_builtin(__builtin_amdgcn_rcpf)
    return __builtin_amdgcn_rcpf(x);
#else
    return 1.0f / x;
#endif
}

// ---------------------------------------------------------------------------
// Kernel 1: centroid fp32 -> bf16 (into d_ws) + u_sq[c] = sum_d centroid^2
// ---------------------------------------------------------------------------
__global__ void centroid_prep_kernel(const float* __restrict__ centroid,
                                     unsigned short* __restrict__ cbf,
                                     float* __restrict__ usq) {
    const int c = blockIdx.x;
    const int t = threadIdx.x;           // 0..127, each handles 4 floats
    __shared__ float ssum;
    if (t == 0) ssum = 0.0f;
    __syncthreads();

    const float4 v = ((const float4*)(centroid + (size_t)c * EMB_DIM))[t];
    float ss = v.x * v.x + v.y * v.y + v.z * v.z + v.w * v.w;

    uint2 packed;
    packed.x = pack_bf16x2(v.x, v.y);
    packed.y = pack_bf16x2(v.z, v.w);
    ((uint2*)(cbf + (size_t)c * EMB_DIM))[t] = packed;

    atomicAdd(&ssum, ss);
    __syncthreads();
    if (t == 0) usq[c] = ssum;
}

// ---------------------------------------------------------------------------
// Kernel 2: block = 256 threads (8 waves), 64 rows x 512 classes.
// Wave w owns columns [64w, 64w+64). B fragments are double-buffered: the
// clause for step kb+1 issues before the 16 WMMAs of step kb, so L2 latency
// is covered by a full k-step of matrix work.
// ---------------------------------------------------------------------------
__global__ void __launch_bounds__(256)
tstudent_assign_kernel(const float* __restrict__ emb,
                       const unsigned short* __restrict__ cbf,
                       const float* __restrict__ usq,
                       float* __restrict__ out) {
    __shared__ alignas(16) unsigned short Alds[M_TILE * ALD_STRIDE];  // ~65 KB
    __shared__ float zsq[M_TILE];
    __shared__ float rowsum[M_TILE];

    const int tid  = threadIdx.x;
    const int row0 = blockIdx.x * M_TILE;

    if (tid < M_TILE) { zsq[tid] = 0.0f; rowsum[tid] = 0.0f; }
    __syncthreads();

    // ---- Stage A: fp32 -> bf16 into padded LDS, accumulate ||z||^2 ----
    {
        const int r  = tid >> 2;               // 0..63 row in tile (4 threads/row)
        const int c0 = (tid & 3) * 128;        // 128-element chunk
        const float* src = emb + (size_t)(row0 + r) * EMB_DIM + c0;
        float ss = 0.0f;
        #pragma unroll
        for (int i = 0; i < 32; ++i) {
            float4 v = ((const float4*)src)[i];
            ss += v.x * v.x + v.y * v.y + v.z * v.z + v.w * v.w;
            uint2 p;
            p.x = pack_bf16x2(v.x, v.y);
            p.y = pack_bf16x2(v.z, v.w);
            *(uint2*)(Alds + r * ALD_STRIDE + c0 + i * 4) = p;
        }
        atomicAdd(&zsq[r], ss);
    }
    __syncthreads();

    // ---- WMMA main loop ----
    const int wid  = tid >> 5;         // wave 0..7 -> columns [64*wid, 64*wid+64)
    const int lane = tid & 31;
    const int half = lane >> 4;        // 0: lanes 0-15, 1: lanes 16-31
    const int nl   = lane & 15;

    const int koffA = half * 16;       // A K split: 0..15 vs 16..31 (contiguous)
    const int koffB = half * 16;       // B K split: 0..15 vs 16..31 (contiguous)

    const v8f vzero = {0.f,0.f,0.f,0.f,0.f,0.f,0.f,0.f};
    v8f acc[4][4];                     // [row tile][col tile]
    #pragma unroll
    for (int rt = 0; rt < 4; ++rt)
        #pragma unroll
        for (int t = 0; t < 4; ++t) acc[rt][t] = vzero;

    union FragU { v16bf v; uint4 q[2]; };

    // Base pointer for this lane's B columns (col = wid*64 + t*16 + nl).
    const unsigned short* bbase =
        cbf + (size_t)(wid * 64 + nl) * EMB_DIM + koffB;

    FragU bfr[2][4];                   // double-buffered B fragments

    // Prologue: load B for kb = 0.
    #pragma unroll
    for (int t = 0; t < 4; ++t) {
        const unsigned short* bp = bbase + (size_t)t * 16 * EMB_DIM;
        bfr[0][t].q[0] = *(const uint4*)(bp);
        bfr[0][t].q[1] = *(const uint4*)(bp + 8);
    }

    #pragma unroll
    for (int kb = 0; kb < EMB_DIM / 32; ++kb) {
        const int cur = kb & 1;
        const int nxt = cur ^ 1;

        // Prefetch B for kb+1 before consuming kb's fragments.
        if (kb + 1 < EMB_DIM / 32) {
            #pragma unroll
            for (int t = 0; t < 4; ++t) {
                const unsigned short* bp =
                    bbase + (size_t)t * 16 * EMB_DIM + (kb + 1) * 32;
                bfr[nxt][t].q[0] = *(const uint4*)(bp);
                bfr[nxt][t].q[1] = *(const uint4*)(bp + 8);
            }
        }

        #pragma unroll
        for (int rt = 0; rt < 4; ++rt) {
            FragU af;                               // low-latency LDS reload per rt
            const unsigned short* ap =
                Alds + (rt * 16 + nl) * ALD_STRIDE + kb * 32 + koffA;
            af.q[0] = *(const uint4*)(ap);
            af.q[1] = *(const uint4*)(ap + 8);
            #pragma unroll
            for (int t = 0; t < 4; ++t)
                acc[rt][t] = __builtin_amdgcn_wmma_f32_16x16x32_bf16(
                    false, af.v, false, bfr[cur][t].v, (short)0, acc[rt][t],
                    false, false);
        }
    }

    // ---- Epilogue: q = 1/(1 + max(z2 - 2*cross + u2, 0)) in-place; row sums ----
    float uq[4];
    #pragma unroll
    for (int t = 0; t < 4; ++t) uq[t] = usq[wid * 64 + t * 16 + nl];

    #pragma unroll
    for (int rt = 0; rt < 4; ++rt) {
        #pragma unroll
        for (int v = 0; v < 8; ++v) {
            const int m = rt * 16 + half * 8 + v;   // C/D layout: M = v + 8*half
            const float zs = zsq[m];
            float part = 0.0f;
            #pragma unroll
            for (int t = 0; t < 4; ++t) {
                float d2 = fmaxf(zs - 2.0f * acc[rt][t][v] + uq[t], 0.0f);
                float q  = fast_rcp(1.0f + d2);     // ALPHA = 1
                acc[rt][t][v] = q;
                part += q;
            }
            // reduce across the 16 lanes of this half (row m lives only here)
            part += __shfl_xor(part, 1, 16);
            part += __shfl_xor(part, 2, 16);
            part += __shfl_xor(part, 4, 16);
            part += __shfl_xor(part, 8, 16);
            if (nl == 0) atomicAdd(&rowsum[m], part);
        }
    }
    __syncthreads();

    // ---- Normalize + store (dense: half-wave writes 64 B contiguous) ----
    #pragma unroll
    for (int rt = 0; rt < 4; ++rt) {
        #pragma unroll
        for (int v = 0; v < 8; ++v) {
            const int m = rt * 16 + half * 8 + v;
            const float inv = fast_rcp(rowsum[m]);
            float* dst = out + (size_t)(row0 + m) * N_CLASS + wid * 64 + nl;
            #pragma unroll
            for (int t = 0; t < 4; ++t)
                dst[t * 16] = acc[rt][t][v] * inv;
        }
    }
}

extern "C" void kernel_launch(void* const* d_in, const int* in_sizes, int n_in,
                              void* d_out, int out_size, void* d_ws, size_t ws_size,
                              hipStream_t stream) {
    const float* emb      = (const float*)d_in[0];   // (262144, 512) f32
    const float* centroid = (const float*)d_in[1];   // (512, 512) f32
    float* out            = (float*)d_out;           // (262144, 512) f32

    unsigned short* cbf = (unsigned short*)d_ws;                         // 512 KB bf16 centroid
    float* usq = (float*)((char*)d_ws + (size_t)N_CLASS * EMB_DIM * 2);  // 2 KB u^2

    centroid_prep_kernel<<<N_CLASS, 128, 0, stream>>>(centroid, cbf, usq);
    tstudent_assign_kernel<<<N_ROWS / M_TILE, 256, 0, stream>>>(emb, cbf, usq, out);
}
